// GNNLSTMPolicy_77154792505459
// MI455X (gfx1250) — matrix operand order repaired and checked by reading, not compiled, not run on hardware
//
#include <hip/hip_runtime.h>
#include <cstdint>
#include <cstddef>

// ---------------------------------------------------------------------------
// GNN-LSTM policy for MI455X (gfx1250, wave32, WMMA).
// LSTM recurrence = per-step GEMM [20000x192]x[192x512] in f16 WMMA
// (v_wmma_f32_16x16x32_f16) with in-register gate fusion (each wave owns all
// four gates of its hidden block, so no LDS spill for the pointwise update).
// Edge phase is bandwidth-bound; native f32 global atomics.
// ---------------------------------------------------------------------------

typedef __attribute__((ext_vector_type(16))) _Float16 v16h;
typedef __attribute__((ext_vector_type(8)))  _Float16 v8h;
typedef __attribute__((ext_vector_type(4)))  _Float16 v4h;
typedef __attribute__((ext_vector_type(8)))  float    v8f;

#define N_   20000
#define T_   32
#define OBS_ 64
#define H_   128
#define A_   8
#define E_   640000
#define G4_  512           // 4*H
#define KC_  192           // OBS + H (fused LSTM GEMM K)

__device__ __forceinline__ float sigf(float x) { return 1.0f / (1.0f + __expf(-x)); }

// ---- WMMA fragment loaders (layouts per CDNA5 ISA 7.12.2) ------

// A: 16x32 f16 tile staged row-major in LDS (stride 32 halves).
// Lane layout: row = lane&15; halves [ksel..ksel+7] and [ksel+16..ksel+23],
// ksel = (lane>=16)*8. Two contiguous 16B chunks -> 2x ds_load_b128.
__device__ __forceinline__ v16h load_a_frag(const _Float16* As, int lane) {
    int arow = lane & 15;
    int ksel = (lane >> 4) * 8;
    const _Float16* p = As + arow * 32 + ksel;
    v8h lo = *(const v8h*)(p);
    v8h hi = *(const v8h*)(p + 16);
    return __builtin_shufflevector(lo, hi, 0, 1, 2, 3, 4, 5, 6, 7,
                                   8, 9, 10, 11, 12, 13, 14, 15);
}

// B: 32x16 tile from row-major f16 weight W[Ncols][ldw]; B[k][n] = W[n][k].
// Per lane: halves [kb+ksel..+7] and [kb+ksel+16..+23], ksel = (lane>=16)*8.
__device__ __forceinline__ v16h load_b_frag(const _Float16* __restrict__ W, int ldw,
                                            int g0, int kb, int lane) {
    int bn = g0 + (lane & 15);
    int kk = kb + ((lane >> 4) * 8);
    const _Float16* p = W + (size_t)bn * ldw + kk;
    v8h lo = *(const v8h*)(p);
    v8h hi = *(const v8h*)(p + 16);
    return __builtin_shufflevector(lo, hi, 0, 1, 2, 3, 4, 5, 6, 7,
                                   8, 9, 10, 11, 12, 13, 14, 15);
}

// ---- setup kernels --------------------------------------------------------

__global__ void k_build_wcat(const float* __restrict__ W_ih, const float* __restrict__ W_hh,
                             const float* __restrict__ b_ih, const float* __restrict__ b_hh,
                             _Float16* __restrict__ Wcat, float* __restrict__ bsum) {
    int idx = blockIdx.x * blockDim.x + threadIdx.x;
    if (idx < G4_ * KC_) {
        int g = idx / KC_, k = idx % KC_;
        float v = (k < OBS_) ? W_ih[g * OBS_ + k] : W_hh[g * H_ + (k - OBS_)];
        Wcat[idx] = (_Float16)v;
    }
    if (idx < G4_) bsum[idx] = b_ih[idx] + b_hh[idx];
}

__global__ void k_conv_f16(const float* __restrict__ src, _Float16* __restrict__ dst, int n) {
    int idx = blockIdx.x * blockDim.x + threadIdx.x;
    if (idx < n) dst[idx] = (_Float16)src[idx];
}

// converts [rows x cols] and zero-pads rows..padRows
__global__ void k_conv_f16_pad(const float* __restrict__ src, _Float16* __restrict__ dst,
                               int rows, int cols, int padRows) {
    int idx = blockIdx.x * blockDim.x + threadIdx.x;
    if (idx >= padRows * cols) return;
    int r = idx / cols;
    dst[idx] = (r < rows) ? (_Float16)src[idx] : (_Float16)0.f;
}

__global__ void k_zero_f32(float* p, int n) {
    int i = blockIdx.x * blockDim.x + threadIdx.x;
    if (i < n) p[i] = 0.f;
}
__global__ void k_zero_f16(_Float16* p, int n) {
    int i = blockIdx.x * blockDim.x + threadIdx.x;
    if (i < n) p[i] = (_Float16)0.f;
}

// ---- fused LSTM step ------------------------------------------------------
// block: 128 threads (4 waves). Tile: 16 rows x 512 gate cols.
// Wave w owns 8 WMMA tiles: (gate g=0..3) x (hidden block b=0..1) at
// columns g*128 + w*32 + b*16. Every lane then holds i/f/g/o for the same
// (row, hidden) in registers -> pointwise update with zero LDS traffic.
__global__ void __launch_bounds__(128)
k_lstm_step(const float* __restrict__ x_seq, const _Float16* __restrict__ Wcat,
            const float* __restrict__ bsum, float* __restrict__ h32,
            float* __restrict__ c32, _Float16* __restrict__ h16, int t) {
    __shared__ __align__(16) _Float16 As[16 * 32];

    const int m0   = blockIdx.x * 16;
    const int lane = threadIdx.x & 31;
    const int wave = threadIdx.x >> 5;
    const int sr   = threadIdx.x >> 3;        // staging row 0..15
    const int sk   = (threadIdx.x & 7) * 4;   // staging k-offset 0..28

    // prefetch h16 tile used from K-iteration 2 onward
    __builtin_prefetch(h16 + (size_t)(m0 + sr) * H_ + sk * 4, 0, 3);

    v8f acc[8];
#pragma unroll
    for (int j = 0; j < 8; ++j) acc[j] = (v8f){};

    for (int kb = 0; kb < KC_; kb += 32) {
        __syncthreads();
        // branch on uniform kb only; per-thread path is one aligned 4-elem copy
        if (kb < OBS_) {
            float4 xv = *(const float4*)(x_seq + (size_t)(m0 + sr) * (T_ * OBS_) +
                                         (size_t)t * OBS_ + kb + sk);
            v4h hv;
            hv[0] = (_Float16)xv.x; hv[1] = (_Float16)xv.y;
            hv[2] = (_Float16)xv.z; hv[3] = (_Float16)xv.w;
            *(v4h*)(As + sr * 32 + sk) = hv;
        } else {
            *(v4h*)(As + sr * 32 + sk) =
                *(const v4h*)(h16 + (size_t)(m0 + sr) * H_ + (kb - OBS_) + sk);
        }
        __syncthreads();
        v16h a = load_a_frag(As, lane);
#pragma unroll
        for (int g = 0; g < 4; ++g)
#pragma unroll
            for (int b = 0; b < 2; ++b) {
                int g0 = g * 128 + wave * 32 + b * 16;
                v16h bfrag = load_b_frag(Wcat, KC_, g0, kb, lane);
                acc[g * 2 + b] = __builtin_amdgcn_wmma_f32_16x16x32_f16(
                    false, a, false, bfrag, (short)0, acc[g * 2 + b], false, false);
            }
    }

    // in-register gate fusion. D layout: M = r + 8*(lane>=16), N = lane&15.
    const int cn = lane & 15, mo = (lane >> 4) * 8;
#pragma unroll
    for (int b = 0; b < 2; ++b) {
        int hid = wave * 32 + b * 16 + cn;
        float bi = bsum[hid], bf = bsum[128 + hid];
        float bg = bsum[256 + hid], bo = bsum[384 + hid];
#pragma unroll
        for (int r = 0; r < 8; ++r) {
            size_t o = (size_t)(m0 + mo + r) * H_ + hid;
            float gi = acc[0 + b][r] + bi;
            float gf = acc[2 + b][r] + bf;
            float gg = acc[4 + b][r] + bg;
            float go = acc[6 + b][r] + bo;
            float cnew = sigf(gf) * c32[o] + sigf(gi) * tanhf(gg);
            float hnew = sigf(go) * tanhf(cnew);
            c32[o] = cnew;
            h32[o] = hnew;
            h16[o] = (_Float16)hnew;
        }
    }
}

// ---- generic WMMA GEMM: Y = [relu](X * Wh^T + bias) -----------------------
// X: f32 [M x K] (ldx, M%16==0, K%32==0), Wh: f16 [NcolsPad x K]
// (NcolsPad%16==0, rows >= Ncols zero-padded), Y: f32 [M x ldy].
// grid: (M/16, ceil(NcolsPad/64)), block 128 (4 waves, 16 cols each).
__global__ void __launch_bounds__(128)
k_gemm16(const float* __restrict__ X, const _Float16* __restrict__ Wh,
         const float* __restrict__ bias, float* __restrict__ Y,
         int M, int K, int Ncols, int NcolsPad, int ldx, int ldy, int relu) {
    __shared__ __align__(16) _Float16 As[16 * 32];
    const int m0   = blockIdx.x * 16;
    const int lane = threadIdx.x & 31;
    const int g0   = blockIdx.y * 64 + (threadIdx.x >> 5) * 16;
    const int sr   = threadIdx.x >> 3;
    const int sk   = (threadIdx.x & 7) * 4;
    const bool act = (g0 < NcolsPad);   // wave-uniform

    v8f acc = (v8f){};
    for (int kb = 0; kb < K; kb += 32) {
        __syncthreads();
        {
            float4 xv = *(const float4*)(X + (size_t)(m0 + sr) * ldx + kb + sk);
            v4h hv;
            hv[0] = (_Float16)xv.x; hv[1] = (_Float16)xv.y;
            hv[2] = (_Float16)xv.z; hv[3] = (_Float16)xv.w;
            *(v4h*)(As + sr * 32 + sk) = hv;
        }
        __syncthreads();
        if (act) {
            v16h a = load_a_frag(As, lane);
            v16h b = load_b_frag(Wh, K, g0, kb, lane);
            acc = __builtin_amdgcn_wmma_f32_16x16x32_f16(false, a, false, b,
                                                         (short)0, acc, false, false);
        }
    }
    const int cn = lane & 15, mo = (lane >> 4) * 8;
    const int gcol = g0 + cn;
    if (gcol < Ncols) {
        float bv = bias ? bias[gcol] : 0.f;
#pragma unroll
        for (int r = 0; r < 8; ++r) {
            float v = acc[r] + bv;
            if (relu) v = fmaxf(v, 0.f);
            Y[(size_t)(m0 + mo + r) * ldy + gcol] = v;
        }
    }
}

// ---- GAT edge phase -------------------------------------------------------

__global__ void k_gat_init(float* __restrict__ agg, unsigned* __restrict__ smax,
                           float* __restrict__ denom) {
    int idx = blockIdx.x * blockDim.x + threadIdx.x;
    if (idx < N_ * H_) agg[idx] = 0.f;
    if (idx < N_) { smax[idx] = 0u; denom[idx] = 0.f; }
}

// wave per edge: s = dot(leaky(xl[src]+xr[dst]+ea*We), att); ordered-uint max.
__global__ void __launch_bounds__(256)
k_edge_score(const float* __restrict__ xl, const float* __restrict__ xr,
             const float* __restrict__ eattr, const float* __restrict__ We,
             const float* __restrict__ att, const int* __restrict__ src,
             const int* __restrict__ dst, float* __restrict__ s_out,
             unsigned* __restrict__ smax) {
    int e    = blockIdx.x * 8 + (threadIdx.x >> 5);
    int lane = threadIdx.x & 31;
    if (e >= E_) return;
    int sN = src[e], dN = dst[e];
    float ea = eattr[e];
    float4 a4 = ((const float4*)(xl + (size_t)sN * H_))[lane];
    float4 b4 = ((const float4*)(xr + (size_t)dN * H_))[lane];
    float4 w4 = ((const float4*)We)[lane];
    float4 t4 = ((const float4*)att)[lane];
    float acc = 0.f;
#define LKDOT(c) { float m = a4.c + b4.c + ea * w4.c; m = (m > 0.f) ? m : 0.2f * m; acc += m * t4.c; }
    LKDOT(x) LKDOT(y) LKDOT(z) LKDOT(w)
#undef LKDOT
    for (int off = 16; off; off >>= 1) acc += __shfl_xor(acc, off, 32);
    if (lane == 0) {
        s_out[e] = acc;
        unsigned bits = __float_as_uint(acc);
        unsigned enc  = (bits & 0x80000000u) ? ~bits : (bits | 0x80000000u);
        atomicMax(&smax[dN], enc);
    }
}

__global__ void k_edge_exp(const float* __restrict__ s, const unsigned* __restrict__ smax,
                           const int* __restrict__ dst, float* __restrict__ exb,
                           float* __restrict__ denom) {
    int e = blockIdx.x * blockDim.x + threadIdx.x;
    if (e >= E_) return;
    int dN = dst[e];
    unsigned enc = smax[dN];
    unsigned bits = (enc & 0x80000000u) ? (enc & 0x7FFFFFFFu) : ~enc;
    float ex = expf(s[e] - __uint_as_float(bits));
    exb[e] = ex;
    atomicAdd(&denom[dN], ex);
}

// wave per edge: agg[dst] += alpha * xl[src]
__global__ void __launch_bounds__(256)
k_edge_scatter(const float* __restrict__ exb, const float* __restrict__ denom,
               const float* __restrict__ xl, const int* __restrict__ src,
               const int* __restrict__ dst, float* __restrict__ agg) {
    int e    = blockIdx.x * 8 + (threadIdx.x >> 5);
    int lane = threadIdx.x & 31;
    if (e >= E_) return;
    int sN = src[e], dN = dst[e];
    float alpha = exb[e] / (denom[dN] + 1e-16f);
    float4 a4 = ((const float4*)(xl + (size_t)sN * H_))[lane];
    float* base = agg + (size_t)dN * H_ + lane * 4;
    atomicAdd(base + 0, alpha * a4.x);
    atomicAdd(base + 1, alpha * a4.y);
    atomicAdd(base + 2, alpha * a4.z);
    atomicAdd(base + 3, alpha * a4.w);
}

// x_out = relu(agg + bias) [+ residual]
__global__ void k_gat_post(const float* __restrict__ agg, const float* __restrict__ bias,
                           const float* __restrict__ res, float* __restrict__ out) {
    int idx = blockIdx.x * blockDim.x + threadIdx.x;
    if (idx >= N_ * H_) return;
    float v = fmaxf(agg[idx] + bias[idx & (H_ - 1)], 0.f);
    if (res) v += res[idx];
    out[idx] = v;
}

// ---- heads ----------------------------------------------------------------

__global__ void __launch_bounds__(256)
k_value(const float* __restrict__ hc, const float* __restrict__ C2w,
        const float* __restrict__ C2b, float* __restrict__ out) {
    int n    = blockIdx.x * 8 + (threadIdx.x >> 5);
    int lane = threadIdx.x & 31;
    if (n >= N_) return;
    float4 h4 = ((const float4*)(hc + (size_t)n * H_))[lane];
    float4 w4 = ((const float4*)C2w)[lane];
    float acc = h4.x * w4.x + h4.y * w4.y + h4.z * w4.z + h4.w * w4.w;
    for (int off = 16; off; off >>= 1) acc += __shfl_xor(acc, off, 32);
    if (lane == 0) out[n] = acc + C2b[0];
}

__global__ void k_std_fill(const float* __restrict__ log_std, float* __restrict__ out) {
    int idx = blockIdx.x * blockDim.x + threadIdx.x;
    if (idx < N_ * A_) out[idx] = expf(log_std[idx & (A_ - 1)]);
}

// ---------------------------------------------------------------------------

extern "C" void kernel_launch(void* const* d_in, const int* in_sizes, int n_in,
                              void* d_out, int out_size, void* d_ws, size_t ws_size,
                              hipStream_t stream) {
    (void)in_sizes; (void)n_in; (void)out_size; (void)ws_size;
    const float* x_seq = (const float*)d_in[0];
    const int*   ei    = (const int*)d_in[1];   // int32 per harness convention
    const float* eatt  = (const float*)d_in[2];
    const float* W_ih  = (const float*)d_in[3];
    const float* W_hh  = (const float*)d_in[4];
    const float* b_ih  = (const float*)d_in[5];
    const float* b_hh  = (const float*)d_in[6];
    const float* Wl1   = (const float*)d_in[7];
    const float* Wr1   = (const float*)d_in[8];
    const float* We1   = (const float*)d_in[9];
    const float* att1  = (const float*)d_in[10];
    const float* bg1   = (const float*)d_in[11];
    const float* Wl2   = (const float*)d_in[12];
    const float* Wr2   = (const float*)d_in[13];
    const float* We2   = (const float*)d_in[14];
    const float* att2  = (const float*)d_in[15];
    const float* bg2   = (const float*)d_in[16];
    const float* A1w   = (const float*)d_in[17];
    const float* A1b   = (const float*)d_in[18];
    const float* A2w   = (const float*)d_in[19];
    const float* A2b   = (const float*)d_in[20];
    const float* C1w   = (const float*)d_in[21];
    const float* C1b   = (const float*)d_in[22];
    const float* C2w   = (const float*)d_in[23];
    const float* C2b   = (const float*)d_in[24];
    const float* lstd  = (const float*)d_in[25];

    const int* src = ei;
    const int* dst = ei + E_;

    // ---- workspace carve-up ----
    char* ws = (char*)d_ws;
    size_t off = 0;
    auto carve = [&](size_t bytes) -> char* {
        off = (off + 255) & ~(size_t)255;
        char* p = ws + off;
        off += bytes;
        return p;
    };
    _Float16* Wcat  = (_Float16*)carve(G4_ * KC_ * 2);
    float*    bsum  = (float*)   carve(G4_ * 4);
    _Float16* Wl1h  = (_Float16*)carve(H_ * H_ * 2);
    _Float16* Wr1h  = (_Float16*)carve(H_ * H_ * 2);
    _Float16* Wl2h  = (_Float16*)carve(H_ * H_ * 2);
    _Float16* Wr2h  = (_Float16*)carve(H_ * H_ * 2);
    _Float16* A1h   = (_Float16*)carve(H_ * H_ * 2);
    _Float16* C1h   = (_Float16*)carve(H_ * H_ * 2);
    _Float16* A2h   = (_Float16*)carve(16 * H_ * 2);  // 8 rows padded to 16
    float*    h32   = (float*)   carve((size_t)N_ * H_ * 4);
    float*    c32   = (float*)   carve((size_t)N_ * H_ * 4);
    _Float16* h16   = (_Float16*)carve((size_t)N_ * H_ * 2);
    float*    xl    = (float*)   carve((size_t)N_ * H_ * 4);
    float*    xr    = (float*)   carve((size_t)N_ * H_ * 4);
    float*    agg   = (float*)   carve((size_t)N_ * H_ * 4);
    float*    x1    = (float*)   carve((size_t)N_ * H_ * 4);
    float*    x2    = (float*)   carve((size_t)N_ * H_ * 4);
    float*    hmlp  = (float*)   carve((size_t)N_ * H_ * 4);
    float*    sbuf  = (float*)   carve((size_t)E_ * 4);
    float*    exb   = (float*)   carve((size_t)E_ * 4);
    unsigned* smax  = (unsigned*)carve((size_t)N_ * 4);
    float*    denom = (float*)   carve((size_t)N_ * 4);

    float* out_mu  = (float*)d_out;
    float* out_std = out_mu + (size_t)N_ * A_;
    float* out_val = out_std + (size_t)N_ * A_;

    const int NB_NH = (N_ * H_ + 255) / 256;
    const int NB_M  = N_ / 16;              // 1250
    const int NB_EW = (E_ + 7) / 8;         // 80000 (wave-per-edge)
    const int NB_ET = (E_ + 255) / 256;     // 2500

    // ---- weight conversion ----
    k_build_wcat<<<(G4_ * KC_ + 255) / 256, 256, 0, stream>>>(W_ih, W_hh, b_ih, b_hh, Wcat, bsum);
    k_conv_f16<<<(H_ * H_ + 255) / 256, 256, 0, stream>>>(Wl1, Wl1h, H_ * H_);
    k_conv_f16<<<(H_ * H_ + 255) / 256, 256, 0, stream>>>(Wr1, Wr1h, H_ * H_);
    k_conv_f16<<<(H_ * H_ + 255) / 256, 256, 0, stream>>>(Wl2, Wl2h, H_ * H_);
    k_conv_f16<<<(H_ * H_ + 255) / 256, 256, 0, stream>>>(Wr2, Wr2h, H_ * H_);
    k_conv_f16<<<(H_ * H_ + 255) / 256, 256, 0, stream>>>(A1w, A1h, H_ * H_);
    k_conv_f16<<<(H_ * H_ + 255) / 256, 256, 0, stream>>>(C1w, C1h, H_ * H_);
    k_conv_f16_pad<<<(16 * H_ + 255) / 256, 256, 0, stream>>>(A2w, A2h, A_, H_, 16);

    // ---- LSTM ----
    k_zero_f32<<<NB_NH, 256, 0, stream>>>(h32, N_ * H_);
    k_zero_f32<<<NB_NH, 256, 0, stream>>>(c32, N_ * H_);
    k_zero_f16<<<NB_NH, 256, 0, stream>>>(h16, N_ * H_);
    for (int t = 0; t < T_; ++t)
        k_lstm_step<<<NB_M, 128, 0, stream>>>(x_seq, Wcat, bsum, h32, c32, h16, t);

    // ---- GAT layer 1 (input h32) ----
    {
        dim3 g(NB_M, 2);
        k_gemm16<<<g, 128, 0, stream>>>(h32, Wl1h, nullptr, xl, N_, H_, H_, H_, H_, H_, 0);
        k_gemm16<<<g, 128, 0, stream>>>(h32, Wr1h, nullptr, xr, N_, H_, H_, H_, H_, H_, 0);
    }
    k_gat_init<<<NB_NH, 256, 0, stream>>>(agg, smax, denom);
    k_edge_score<<<NB_EW, 256, 0, stream>>>(xl, xr, eatt, We1, att1, src, dst, sbuf, smax);
    k_edge_exp<<<NB_ET, 256, 0, stream>>>(sbuf, smax, dst, exb, denom);
    k_edge_scatter<<<NB_EW, 256, 0, stream>>>(exb, denom, xl, src, dst, agg);
    k_gat_post<<<NB_NH, 256, 0, stream>>>(agg, bg1, nullptr, x1);

    // ---- GAT layer 2 (input x1, residual x1) ----
    {
        dim3 g(NB_M, 2);
        k_gemm16<<<g, 128, 0, stream>>>(x1, Wl2h, nullptr, xl, N_, H_, H_, H_, H_, H_, 0);
        k_gemm16<<<g, 128, 0, stream>>>(x1, Wr2h, nullptr, xr, N_, H_, H_, H_, H_, H_, 0);
    }
    k_gat_init<<<NB_NH, 256, 0, stream>>>(agg, smax, denom);
    k_edge_score<<<NB_EW, 256, 0, stream>>>(xl, xr, eatt, We2, att2, src, dst, sbuf, smax);
    k_edge_exp<<<NB_ET, 256, 0, stream>>>(sbuf, smax, dst, exb, denom);
    k_edge_scatter<<<NB_EW, 256, 0, stream>>>(exb, denom, xl, src, dst, agg);
    k_gat_post<<<NB_NH, 256, 0, stream>>>(agg, bg2, x1, x2);

    // ---- actor head ----
    {
        dim3 g(NB_M, 2);
        k_gemm16<<<g, 128, 0, stream>>>(x2, A1h, A1b, hmlp, N_, H_, H_, H_, H_, H_, 1);
        dim3 g2(NB_M, 1);
        k_gemm16<<<g2, 128, 0, stream>>>(hmlp, A2h, A2b, out_mu, N_, H_, A_, 16, H_, A_, 0);
    }
    k_std_fill<<<(N_ * A_ + 255) / 256, 256, 0, stream>>>(lstd, out_std);

    // ---- critic head ----
    {
        dim3 g(NB_M, 2);
        k_gemm16<<<g, 128, 0, stream>>>(x2, C1h, C1b, hmlp, N_, H_, H_, H_, H_, H_, 1);
    }
    k_value<<<(N_ + 7) / 8, 256, 0, stream>>>(hmlp, C2w, C2b, out_val);
}
